// VectorQuantizer_53901839565722
// MI455X (gfx1250) — matrix operand order repaired and checked by reading, not compiled, not run on hardware
//
#include <hip/hip_runtime.h>

#define D 128

typedef __attribute__((ext_vector_type(16))) __bf16 v16bf;
typedef __attribute__((ext_vector_type(8)))  __bf16 v8bf;
typedef __attribute__((ext_vector_type(4)))  __bf16 v4bf;
typedef __attribute__((ext_vector_type(8)))  float  v8f;

// ---------------------------------------------------------------------------
// Prep: one wave per row. Convert a 128-float row to bf16 (coalesced: each
// lane handles 4 consecutive floats = 16B load, 8B store). Optionally compute
// the exact f32 row sum-of-squares (codebook ||e||^2).
// ---------------------------------------------------------------------------
__global__ void vq_prep(const float* __restrict__ src, __bf16* __restrict__ dst,
                        float* __restrict__ b2, int rows) {
  const int wave = (blockIdx.x * blockDim.x + threadIdx.x) >> 5;
  const int lane = threadIdx.x & 31;
  if (wave >= rows) return;
  const float4 v = *((const float4*)(src + (size_t)wave * D) + lane);
  v4bf o;
  o[0] = (__bf16)v.x; o[1] = (__bf16)v.y; o[2] = (__bf16)v.z; o[3] = (__bf16)v.w;
  *(v4bf*)(dst + (size_t)wave * D + lane * 4) = o;
  if (b2 != nullptr) {
    float ss = v.x * v.x + v.y * v.y + v.z * v.z + v.w * v.w;
    #pragma unroll
    for (int off = 16; off > 0; off >>= 1) ss += __shfl_xor(ss, off, 32);
    if (lane == 0) b2[wave] = ss;
  }
}

__global__ void vq_zero(unsigned* __restrict__ p, int n) {
  int i = blockIdx.x * blockDim.x + threadIdx.x;
  if (i < n) p[i] = 0u;
}

// ---------------------------------------------------------------------------
// Fused GEMM + argmin, software-pipelined 2 tiles deep. 8 waves / block, each
// wave owns a 16-row M-tile (block = 128 rows) and sweeps all V codes in
// 16-wide N-tiles. A fragments (16x128 bf16) live in registers for the whole
// sweep. Per iteration: issue loads for tile t+1, run WMMA chain for tile t,
// issue loads for tile t+2, run WMMA chain for tile t+1, THEN both argmin
// epilogues. The two back-to-back WMMA chains fill each other's WMMA->VALU
// hazard slots and hide the B-load latency behind matrix work.
// dist = ||e||^2 - 4*(x.e); +||x||^2 is row-constant, cannot change argmin.
// C/D layout: lane L holds col n = L%16, rows M = 8*(L/16)+i for VGPR i.
// Ebf/b2 carry one 16-row padding tile so the pipelined loads stay branchless.
// ---------------------------------------------------------------------------
__global__ void __launch_bounds__(256)
vq_gemm_argmin(const __bf16* __restrict__ Xbf, const __bf16* __restrict__ Ebf,
               const float* __restrict__ b2, int* __restrict__ min_idx, int V) {
  const int lane = threadIdx.x & 31;
  const int wave = threadIdx.x >> 5;
  const int rowBase = blockIdx.x * 128 + wave * 16;
  const int half = lane >> 4;     // which 16-lane half of the wave
  const int l16  = lane & 15;

  // A fragment: lane (M=l16, half) holds k = 16*half + [0,16) contiguous
  // within each 32-wide K tile.
  v16bf A[4];
  const __bf16* xrow = Xbf + (size_t)(rowBase + l16) * D;
  #pragma unroll
  for (int kt = 0; kt < 4; ++kt) {
    const int kb = kt * 32;
    v8bf lo = *(const v8bf*)(xrow + kb + 16 * half);
    v8bf hi = *(const v8bf*)(xrow + kb + 16 * half + 8);
    #pragma unroll
    for (int i = 0; i < 8; ++i) { A[kt][i] = lo[i]; A[kt][8 + i] = hi[i]; }
  }

  float run_d[8];
  int   run_i[8];
  #pragma unroll
  for (int i = 0; i < 8; ++i) { run_d[i] = 3.4e38f; run_i[i] = 0; }

  // B fragment: lane (n=l16, half) holds k = 16*half + [0,16) contiguous.
  const size_t tileStep = (size_t)16 * D;            // elements per 16-code tile
  const __bf16* ep = Ebf + (size_t)l16 * D + 16 * half;
  const float*  bp = b2 + l16;

  v16bf Ba[4], Bb[4];
  float sa, sb;
  #pragma unroll
  for (int kt = 0; kt < 4; ++kt) Ba[kt] = *(const v16bf*)(ep + kt * 32);
  sa = bp[0];

  for (int nt = 0; nt < V; nt += 32) {
    // loads for tile nt+16
    const __bf16* ep1 = ep + tileStep;
    #pragma unroll
    for (int kt = 0; kt < 4; ++kt) Bb[kt] = *(const v16bf*)(ep1 + kt * 32);
    sb = bp[nt + 16];

    // WMMA chain, tile nt (Ba loaded one iteration ago)
    v8f acc_a = {};
    #pragma unroll
    for (int kt = 0; kt < 4; ++kt)
      acc_a = __builtin_amdgcn_wmma_f32_16x16x32_bf16(
          false, A[kt], false, Ba[kt], (short)0, acc_a, false, false);

    // loads for tile nt+32 (last round reads the padding tile; never consumed)
    const __bf16* ep2 = ep + 2 * tileStep;
    #pragma unroll
    for (int kt = 0; kt < 4; ++kt) Ba[kt] = *(const v16bf*)(ep2 + kt * 32);
    const float sa_next = bp[nt + 32];

    // WMMA chain, tile nt+16 (independent of acc_a: fills its hazard slots)
    v8f acc_b = {};
    #pragma unroll
    for (int kt = 0; kt < 4; ++kt)
      acc_b = __builtin_amdgcn_wmma_f32_16x16x32_bf16(
          false, A[kt], false, Bb[kt], (short)0, acc_b, false, false);

    // argmin epilogues (acc_a's VALU covers acc_b's WMMA->VALU hazard)
    {
      const int code = nt + l16;
      #pragma unroll
      for (int i = 0; i < 8; ++i) {
        float d = __builtin_fmaf(acc_a[i], -4.0f, sa);
        bool c = d < run_d[i];             // strict <: lane's codes increase,
        run_d[i] = c ? d : run_d[i];       // so ties keep the earliest index
        run_i[i] = c ? code : run_i[i];
      }
    }
    {
      const int code = nt + 16 + l16;
      #pragma unroll
      for (int i = 0; i < 8; ++i) {
        float d = __builtin_fmaf(acc_b[i], -4.0f, sb);
        bool c = d < run_d[i];
        run_d[i] = c ? d : run_d[i];
        run_i[i] = c ? code : run_i[i];
      }
    }

    sa = sa_next;
    ep += 2 * tileStep;
  }

  // Reduce across the 16 lanes of each half (masks < 16 stay in the half).
  #pragma unroll
  for (int off = 1; off < 16; off <<= 1) {
    #pragma unroll
    for (int i = 0; i < 8; ++i) {
      float od = __shfl_xor(run_d[i], off, 32);
      int   oi = __shfl_xor(run_i[i], off, 32);
      bool take = (od < run_d[i]) || (od == run_d[i] && oi < run_i[i]);
      run_d[i] = take ? od : run_d[i];
      run_i[i] = take ? oi : run_i[i];
    }
  }
  if (l16 == 0) {
    #pragma unroll
    for (int i = 0; i < 8; ++i) min_idx[rowBase + half * 8 + i] = run_i[i];
  }
}

// ---------------------------------------------------------------------------
// Gather codebook rows (exact f32), emit quantized_st and enc_idx, per-row
// squared error, and the code histogram (integer atomics: deterministic).
// ---------------------------------------------------------------------------
__global__ void vq_finalize(const float* __restrict__ x, const float* __restrict__ emb,
                            const int* __restrict__ min_idx,
                            float* __restrict__ out_q, float* __restrict__ out_idx,
                            float* __restrict__ row_sse, unsigned* __restrict__ counts) {
  __shared__ float red[128];
  const int r = blockIdx.x, t = threadIdx.x;
  const int idx = min_idx[r];
  const float xv = x[(size_t)r * D + t];
  const float qv = emb[(size_t)idx * D + t];
  out_q[(size_t)r * D + t] = xv + (qv - xv);   // straight-through, literally
  const float dif = qv - xv;
  red[t] = dif * dif;
  __syncthreads();
  #pragma unroll
  for (int s = 64; s > 0; s >>= 1) {
    if (t < s) red[t] += red[t + s];
    __syncthreads();
  }
  if (t == 0) {
    row_sse[r] = red[0];
    out_idx[r] = (float)idx;
    atomicAdd(&counts[idx], 1u);
  }
}

// Single-block deterministic reductions: loss = 1.25 * mse, perplexity.
__global__ void vq_reduce(const float* __restrict__ row_sse,
                          const unsigned* __restrict__ counts,
                          float* __restrict__ out_loss, float* __restrict__ out_ppl,
                          int N, int V) {
  __shared__ float s1[256], s2[256];
  const int t = threadIdx.x;
  float a = 0.f, b = 0.f;
  for (int i = t; i < N; i += 256) a += row_sse[i];
  const float invN = 1.0f / (float)N;
  for (int v = t; v < V; v += 256) {
    float p = (float)counts[v] * invN;
    b += p * logf(p + 1e-10f);
  }
  s1[t] = a; s2[t] = b;
  __syncthreads();
  #pragma unroll
  for (int s = 128; s > 0; s >>= 1) {
    if (t < s) { s1[t] += s1[t + s]; s2[t] += s2[t + s]; }
    __syncthreads();
  }
  if (t == 0) {
    float mse = s1[0] / (float)((size_t)N * D);
    *out_loss = mse * 1.25f;          // q_latent + COMMIT*e_latent, identical terms
    *out_ppl  = expf(-s2[0]);
  }
}

extern "C" void kernel_launch(void* const* d_in, const int* in_sizes, int n_in,
                              void* d_out, int out_size, void* d_ws, size_t ws_size,
                              hipStream_t stream) {
  const float* x   = (const float*)d_in[0];   // [8,2048,128] f32
  const float* emb = (const float*)d_in[1];   // [8192,128]  f32
  const int N = in_sizes[0] / D;              // 16384
  const int V = in_sizes[1] / D;              // 8192

  char* ws = (char*)d_ws;
  size_t off = 0;
  __bf16* Xbf = (__bf16*)(ws + off); off += (size_t)N * D * sizeof(__bf16);
  // +16 rows of padding: the pipelined B prefetch over-reads one tile
  __bf16* Ebf = (__bf16*)(ws + off); off += (size_t)(V + 16) * D * sizeof(__bf16);
  off = (off + 255) & ~(size_t)255;
  float*    b2      = (float*)(ws + off);    off += (size_t)(V + 16) * sizeof(float);
  int*      min_idx = (int*)(ws + off);      off += (size_t)N * sizeof(int);
  float*    row_sse = (float*)(ws + off);    off += (size_t)N * sizeof(float);
  unsigned* counts  = (unsigned*)(ws + off); off += (size_t)V * sizeof(unsigned);

  float* out_q    = (float*)d_out;            // [N*D]
  float* out_idx  = out_q + (size_t)N * D;    // [N]
  float* out_loss = out_idx + N;              // [1]
  float* out_ppl  = out_loss + 1;             // [1]

  vq_zero<<<(V + 255) / 256, 256, 0, stream>>>(counts, V);
  vq_prep<<<(N * 32 + 255) / 256, 256, 0, stream>>>(x, Xbf, nullptr, N);
  vq_prep<<<(V * 32 + 255) / 256, 256, 0, stream>>>(emb, Ebf, b2, V);
  vq_gemm_argmin<<<N / 128, 256, 0, stream>>>(Xbf, Ebf, b2, min_idx, V);
  vq_finalize<<<N, 128, 0, stream>>>(x, emb, min_idx, out_q, out_idx, row_sse, counts);
  vq_reduce<<<1, 256, 0, stream>>>(row_sse, counts, out_loss, out_ppl, N, V);
}
